// Base_RNN_16527034155630
// MI455X (gfx1250) — matrix operand order repaired and checked by reading, not compiled, not run on hardware
//
#include <hip/hip_runtime.h>

// ---------------------------------------------------------------------------
// tanh-RNN, H=4096: 1024 teacher-forced steps + 128 AR steps, all sequential.
// Persistent-grid register-resident RNN:
//   - Vh packed once to bf16 WMMA A-fragment layout (pack kernel).
//   - Each wave preloads its 32 A-fragments (256 VGPRs, CDNA5 extended VGPR
//     file) BEFORE the time loop -> whole 32MB weight matrix lives in the
//     chip's register files across all 1152 steps; zero weight streaming.
//   - Per step: stage h (bf16, 8KB) to LDS, 32 x v_wmma_f32_16x16x32_bf16.
//     B-fragment ds_loads software-pipelined 8 deep via sched_group_barrier
//     so LDS latency hides behind WMMA issue (counted s_wait_dscnt).
//   - amdgpu_waves_per_eu(2): 8 waves/WGP target -> scheduler gets a large
//     VGPR budget and stops collapsing the pipeline buffers.
//   - Grid-wide step barrier: monotonic device-scope atomic + s_sleep.
// ---------------------------------------------------------------------------

#define H        4096
#define TSEQ     1024
#define HORIZON  128
#define NBLK     128
#define NTHR     256                      // 8 wave32 per block
#define ROWS_PER_BLK   (H / NBLK)         // 32
#define TILES_PER_BLK  (ROWS_PER_BLK/16)  // 2 row tiles of 16
#define KSLICES        4                  // K split across wave pairs
#define KCHUNKS        (H / 32)           // 128 chunks of K=32
#define KCHUNKS_SLICE  (KCHUNKS / KSLICES)// 32 chunks per wave

#define PACK_BYTES ((size_t)H * H * 2)    // bf16 Vh, fragment layout: 32MB
#define HBF_BYTES  ((size_t)2 * H * 2)    // double-buffered bf16 h
#define HF_BYTES   ((size_t)2 * H * 4)    // double-buffered f32  h

typedef __attribute__((ext_vector_type(16))) __bf16 v16bf;
typedef __attribute__((ext_vector_type(8)))  float  v8f;

#define SGB __builtin_amdgcn_sched_group_barrier
#define SG_MFMA  0x008   // MFMA/WMMA group
#define SG_DSRD  0x100   // DS-read group

__device__ __forceinline__ unsigned short f2bf(float f) {
  unsigned u = __float_as_uint(f);
  u += 0x7FFFu + ((u >> 16) & 1u);        // round-to-nearest-even
  return (unsigned short)(u >> 16);
}

// ---------------------------------------------------------------------------
// One-time pack: Vh (= V[:,1:]) f32 -> bf16 in WMMA A-fragment order.
// A 16x32 bf16, contiguous k used here (not ISA table order): lane l(0-15)=row l, K = 0..15;
// lane l+16 = row l, K = 16..31. One thread per (fragment, lane).
// ---------------------------------------------------------------------------
__global__ void pack_vh_kernel(const float* __restrict__ V,
                               unsigned short* __restrict__ packA) {
  int gid  = blockIdx.x * blockDim.x + threadIdx.x;  // 256*128*32 threads
  int lane = gid & 31;
  int frag = gid >> 5;                // rt*KCHUNKS + kc
  int kc   = frag & (KCHUNKS - 1);
  int rt   = frag >> 7;
  int grp  = lane >> 4;
  int row  = rt * 16 + (lane & 15);
  unsigned short* dst = packA + ((size_t)frag * 32 + lane) * 16;
  const float* src = V + (size_t)row * (H + 1) + 1 + (size_t)kc * 32;
#pragma unroll
  for (int j = 0; j < 16; ++j) {
    int k = grp * 16 + j;
    dst[j] = f2bf(src[k]);
  }
}

__global__ void init_state_kernel(unsigned short* hbf, float* hf32,
                                  unsigned* cnt) {
  int i = blockIdx.x * blockDim.x + threadIdx.x;
  if (i == 0) *cnt = 0u;
  if (i < 2 * H) { hbf[i] = 0; hf32[i] = 0.f; }
}

// ---------------------------------------------------------------------------
// Persistent RNN kernel.
// ---------------------------------------------------------------------------
__global__ __launch_bounds__(NTHR)
__attribute__((amdgpu_waves_per_eu(2)))
void rnn_persistent_kernel(
    const float* __restrict__ xseq, const float* __restrict__ V,
    const float* __restrict__ bvec, const float* __restrict__ W,
    const float* __restrict__ cptr, const unsigned short* __restrict__ packA,
    unsigned short* __restrict__ hbf, float* __restrict__ hf32,
    unsigned* __restrict__ barrier_cnt, float* __restrict__ out) {
  __shared__ alignas(32) unsigned short sh_h[H];               // 8KB bf16 h
  __shared__ float sh_part[TILES_PER_BLK][16][KSLICES];        // wmma partials
  __shared__ float sh_red[NTHR];
  __shared__ float sh_y;

  const int tid    = threadIdx.x;
  const int lane   = tid & 31;
  const int wid    = tid >> 5;
  const int tile   = wid & 1;        // which 16-row tile
  const int kslice = wid >> 1;       // which quarter of K
  const int grp    = lane >> 4;      // lane half
  const int rtG    = blockIdx.x * TILES_PER_BLK + tile;
  const v16bf* __restrict__ Afrags = (const v16bf*)packA;

  // -------- Preload this wave's 32 weight fragments into registers --------
  v16bf areg[KCHUNKS_SLICE];
  {
    const size_t fragBase =
        ((size_t)rtG * KCHUNKS + (size_t)kslice * KCHUNKS_SLICE) * 32 + lane;
#pragma unroll
    for (int cc = 0; cc < KCHUNKS_SLICE; ++cc)
      areg[cc] = Afrags[fragBase + (size_t)cc * 32];
  }
  // LDS base for this wave's B fragments (chunk cc of its K slice)
  const unsigned short* hslice = &sh_h[(kslice * KCHUNKS_SLICE) * 32 + grp * 16];

  float yloc = 0.f;

  for (int t = 0; t < TSEQ + HORIZON; ++t) {
    const int rbuf = (t & 1) ^ 1;    // h_{t-1}
    const int wbuf = t & 1;          // h_t
    const float x = (t < TSEQ) ? xseq[t] : yloc;

    // Stage previous h (bf16) into LDS for B fragments (8KB, broadcast reads)
    {
      const uint4* src = (const uint4*)(hbf + (size_t)rbuf * H);
      uint4* dst = (uint4*)sh_h;
      for (int i = tid; i < (H * 2) / 16; i += NTHR) dst[i] = src[i];
    }
    __syncthreads();

    // --- WMMA matvec: C[m][n] = sum_k Vh[m][k] * h[k], h broadcast over n ---
    // B 32x16 bf16, same contiguous k order as A: lanes 0-15 hold K=0..15, lanes 16-31 hold K=16..31.
    v8f acc = {0.f, 0.f, 0.f, 0.f, 0.f, 0.f, 0.f, 0.f};
#pragma unroll
    for (int cc = 0; cc < KCHUNKS_SLICE; ++cc) {
      v16bf bfc = *(const v16bf*)(hslice + cc * 32);
      acc = __builtin_amdgcn_wmma_f32_16x16x32_bf16(
          false, areg[cc], false, bfc, (short)0, acc, false, false);
    }
    // Pin an 8-deep software pipeline on this block's schedule:
    // 16 ds_reads ahead, then [1 wmma, 2 ds_reads] x24, then 8 wmmas.
    SGB(SG_DSRD, 16, 0);
#pragma unroll
    for (int i = 0; i < 24; ++i) { SGB(SG_MFMA, 1, 0); SGB(SG_DSRD, 2, 0); }
    SGB(SG_MFMA, 8, 0);

    // C/D layout: lane n(0-15) VGPR r = C[M=r][n]; lane n(16-31) = C[M=8+r].
    // All columns identical (B broadcast) -> lanes 0 and 16 carry the rows.
    if ((lane & 15) == 0) {
#pragma unroll
      for (int r = 0; r < 8; ++r) sh_part[tile][grp * 8 + r][kslice] = acc[r];
    }
    __syncthreads();

    // --- h2 = tanh(Vx*x + Vh@h + b) for this block's 32 rows ---
    if (tid < ROWS_PER_BLK) {
      const int tl = tid >> 4, rr = tid & 15;
      const int grow = blockIdx.x * ROWS_PER_BLK + tid;
      float z = bvec[grow] + V[(size_t)grow * (H + 1)] * x;
#pragma unroll
      for (int s = 0; s < KSLICES; ++s) z += sh_part[tl][rr][s];
      float h2 = tanhf(z);
      hf32[(size_t)wbuf * H + grow] = h2;
      hbf[(size_t)wbuf * H + grow]  = f2bf(h2);
    }

    // --- grid barrier (release h_t, acquire everyone else's h_t) ---
    __threadfence();                  // each thread flushes its own stores
    __syncthreads();
    if (tid == 0) {
      __hip_atomic_fetch_add(barrier_cnt, 1u, __ATOMIC_RELEASE,
                             __HIP_MEMORY_SCOPE_AGENT);
      const unsigned target = (unsigned)(t + 1) * (unsigned)NBLK;
      while (__hip_atomic_load(barrier_cnt, __ATOMIC_ACQUIRE,
                               __HIP_MEMORY_SCOPE_AGENT) < target)
        __builtin_amdgcn_s_sleep(1);
    }
    __syncthreads();
    __threadfence();                  // invalidate stale lines before reads

    // --- y_t = W . h_t + c (redundant per block, deterministic order) ---
    if (t >= TSEQ - 1) {
      const float* hcur = hf32 + (size_t)wbuf * H;
      float p = 0.f;
      for (int i = tid; i < H; i += NTHR) p += W[i] * hcur[i];
      sh_red[tid] = p;
      __syncthreads();
      if (tid == 0) {
        float s = cptr[0];
        for (int i = 0; i < NTHR; ++i) s += sh_red[i];
        sh_y = s;
        if (t >= TSEQ && blockIdx.x == 0) out[t - TSEQ] = s;
      }
      __syncthreads();
      yloc = sh_y;
    }
  }
}

// ---------------------------------------------------------------------------
extern "C" void kernel_launch(void* const* d_in, const int* in_sizes, int n_in,
                              void* d_out, int out_size, void* d_ws,
                              size_t ws_size, hipStream_t stream) {
  const float* xseq = (const float*)d_in[0];
  const float* V    = (const float*)d_in[1];
  const float* b    = (const float*)d_in[2];
  const float* W    = (const float*)d_in[3];
  const float* c    = (const float*)d_in[4];
  float* out = (float*)d_out;

  char* ws = (char*)d_ws;
  unsigned short* packA = (unsigned short*)ws;
  unsigned short* hbf   = (unsigned short*)(ws + PACK_BYTES);
  float*          hf32  = (float*)(ws + PACK_BYTES + HBF_BYTES);
  unsigned*       cnt   = (unsigned*)(ws + PACK_BYTES + HBF_BYTES + HF_BYTES);

  // (H/16)*(H/32) fragments * 32 lanes = 1,048,576 pack threads
  pack_vh_kernel<<<4096, 256, 0, stream>>>(V, packA);
  init_state_kernel<<<32, 256, 0, stream>>>(hbf, hf32, cnt);
  rnn_persistent_kernel<<<NBLK, NTHR, 0, stream>>>(
      xseq, V, b, W, c, packA, hbf, hf32, cnt, out);
}
